// Region_Proposal_Network_42090679500961
// MI455X (gfx1250) — matrix operand order repaired and checked by reading, not compiled, not run on hardware
//
#include <hip/hip_runtime.h>
#include <hip/hip_bf16.h>
#include <math.h>

typedef __attribute__((ext_vector_type(16))) _Float16 v16h;
typedef __attribute__((ext_vector_type(8)))  _Float16 v8h;
typedef __attribute__((ext_vector_type(8)))  float    v8f;
typedef __attribute__((ext_vector_type(4)))  unsigned int u32x4;

#define NIMG   4
#define HW     64
#define CIN    512
#define COUT   512
#define KDIM   4608            // 9 * 512
#define PADW   66
#define NPIX   4096            // 64*64
#define NANCH  36864           // 4096 * 9
#define NPOST  3000

// ---- output layout (float elements, concatenated reference outputs) ----
constexpr size_t OFF_LOCS   = 0;                       // [4][36864][4]
constexpr size_t OFF_SCORES = 589824;                  // [4][36864][2]
constexpr size_t OFF_ROIS   = 884736;                  // [12000][4]
constexpr size_t OFF_RIDX   = 932736;                  // [12000]
constexpr size_t OFF_ANCH   = 944736;                  // [36864][4]

// ---- workspace layout (bytes) ----
constexpr size_t WS_PADIN  = 0;                        // f16 [4][66][66][512]
constexpr size_t WS_BT     = 17842176;                 // f16 [512][4608]
constexpr size_t WS_FM     = 22560768;                 // f16 [4][64][64][512]
constexpr size_t WS_WT     = 39337984;                 // f16 [64][512]
constexpr size_t WS_HB     = 39403520;                 // f32 [64]
constexpr size_t WS_BOXES  = 39403776;                 // f32 [4][36864][4]
constexpr size_t WS_SCR    = 41763072;                 // f32 [4][36864]

// CDNA5 async global->LDS copy (ASYNCcnt-tracked), 16 B per lane.
// LDS destination address = low 32 bits of the generic pointer.
__device__ __forceinline__ void async_copy_b128(void* lds_dst, const void* gsrc)
{
  unsigned lds = (unsigned)(unsigned long long)lds_dst;
  asm volatile("global_load_async_to_lds_b128 %0, %1, off"
               :: "v"(lds), "v"(gsrc) : "memory");
}
__device__ __forceinline__ void wait_async0()
{
  asm volatile("s_wait_asynccnt 0" ::: "memory");
}

// =====================================================================
// Prep: NCHW f32 image -> zero-padded NHWC f16
// =====================================================================
__global__ __launch_bounds__(256) void pad_input_kernel(
    const float* __restrict__ img, _Float16* __restrict__ padIn)
{
  int idx = blockIdx.x * 256 + threadIdx.x;
  if (idx >= NIMG * PADW * PADW * CIN) return;
  int c  = idx & 511;
  int xx = (idx >> 9) % PADW;
  int yy = (idx >> 9) / PADW % PADW;
  int n  = idx / (512 * PADW * PADW);
  float v = 0.f;
  if (yy >= 1 && yy <= 64 && xx >= 1 && xx <= 64)
    v = img[(((size_t)n * CIN + c) * HW + (yy - 1)) * HW + (xx - 1)];
  padIn[idx] = (_Float16)v;
}

// conv_w [o][c][ky][kx] f32 -> Bt[o][(ky*3+kx)*512 + c] f16
__global__ __launch_bounds__(256) void prep_w3_kernel(
    const float* __restrict__ wsrc, _Float16* __restrict__ Bt)
{
  int idx = blockIdx.x * 256 + threadIdx.x;
  if (idx >= COUT * KDIM) return;
  int o = idx / KDIM, k = idx % KDIM;
  int tap = k >> 9, c = k & 511;
  int ky = tap / 3, kx = tap % 3;
  Bt[idx] = (_Float16)wsrc[(((size_t)o * CIN + c) * 3 + ky) * 3 + kx];
}

// head weights: o<36 -> reg_w, 36<=o<54 -> cls_w, else 0 ; plus bias vector
__global__ __launch_bounds__(256) void prep_whead_kernel(
    const float* __restrict__ cls_w, const float* __restrict__ cls_b,
    const float* __restrict__ reg_w, const float* __restrict__ reg_b,
    _Float16* __restrict__ Wt, float* __restrict__ hbias)
{
  int idx = blockIdx.x * 256 + threadIdx.x;
  if (idx < 64 * CIN) {
    int o = idx >> 9, c = idx & 511;
    float v = 0.f;
    if (o < 36)      v = reg_w[(size_t)o * CIN + c];
    else if (o < 54) v = cls_w[(size_t)(o - 36) * CIN + c];
    Wt[idx] = (_Float16)v;
  }
  if (idx < 64) {
    float b = 0.f;
    if (idx < 36)      b = reg_b[idx];
    else if (idx < 54) b = cls_b[idx - 36];
    hbias[idx] = b;
  }
}

// =====================================================================
// 3x3 conv as implicit GEMM with WMMA f32<=f16.
// Double-buffered LDS staging filled by global_load_async_to_lds_b128;
// one barrier per K-step, async prefetch of stage k+1 overlaps WMMAs of k.
// grid = 1024 blocks: blk&3 = 128-channel tile, blk>>2 = (n,y) pixel row
// =====================================================================
__global__ __launch_bounds__(256) void conv3x3_wmma(
    const _Float16* __restrict__ padIn, const _Float16* __restrict__ Bt,
    const float* __restrict__ conv_b, _Float16* __restrict__ fm)
{
  __shared__ _Float16 As[2][64][32];    // 2 x 4 KB
  __shared__ _Float16 Bs[2][128][32];   // 2 x 8 KB

  int blk   = blockIdx.x;
  int ntile = blk & 3;
  int mrow  = blk >> 2;              // 0..255
  int n = mrow >> 6, y = mrow & 63;
  int t = threadIdx.x;
  int w = t >> 5, lane = t & 31;
  int msub  = (w & 3) * 16;          // wave's pixel sub-tile
  int nbase = (w >> 2) * 64;         // wave's column group (0 or 64)

  v8f acc[4] = {};

  int apix = t >> 2, aseg = t & 3;   // A loader: 16 B per thread
  int brow = t >> 1, bseg = t & 1;   // B loader: 32 B per thread
  int obrow = ntile * 128 + brow;

  auto issue_stage = [&](int ks, int p) {
    int tap  = ks >> 4;              // (ky,kx) tap, 512/32 = 16 steps per tap
    int cblk = (ks & 15) * 32;
    int ky = tap / 3, kx = tap % 3;
    const _Float16* ga = padIn +
        (((size_t)(n * PADW + y + ky) * PADW + (apix + kx)) * CIN + cblk) + aseg * 8;
    async_copy_b128(&As[p][apix][aseg * 8], ga);
    const _Float16* gb = Bt + (size_t)obrow * KDIM + ks * 32 + bseg * 16;
    async_copy_b128(&Bs[p][brow][bseg * 16], gb);
    async_copy_b128(&Bs[p][brow][bseg * 16 + 8], gb + 8);
  };

  issue_stage(0, 0);

  for (int ks = 0; ks < 144; ++ks) {
    int p = ks & 1;
    wait_async0();                   // our stage-ks transfers have landed in LDS
    __syncthreads();                 // everyone's have
    if (ks + 1 < 144) issue_stage(ks + 1, p ^ 1);   // prefetch overlaps compute

    // A fragment: lanes 0-15 => K 0-7 / 16-23 ; lanes 16-31 => K 8-15 / 24-31
    union { v16h v; v8h h[2]; } af;
    int am = msub + (lane & 15);
    int ko = (lane >> 4) * 8;
    af.h[0] = *(const v8h*)&As[p][am][ko];
    af.h[1] = *(const v8h*)&As[p][am][16 + ko];

#pragma unroll
    for (int tc = 0; tc < 4; ++tc) {
      union { v16h v; v8h h[2]; } bf;   // column per lane, K 0-7/16-23 (lo) 8-15/24-31 (hi)
      int col = nbase + tc * 16 + (lane & 15);
      int bko = (lane >> 4) * 8;
      bf.h[0] = *(const v8h*)&Bs[p][col][bko];
      bf.h[1] = *(const v8h*)&Bs[p][col][bko + 16];
      acc[tc] = __builtin_amdgcn_wmma_f32_16x16x32_f16(
          false, af.v, false, bf.v, (short)0, acc[tc], false, false);
    }
  }

#pragma unroll
  for (int tc = 0; tc < 4; ++tc) {
    int o = ntile * 128 + nbase + tc * 16 + (lane & 15);
    float bias = conv_b[o];
#pragma unroll
    for (int r = 0; r < 8; ++r) {
      int x = msub + (lane >> 4) * 8 + r;    // C/D layout: M = r (+8 for hi lanes)
      float v = acc[tc][r] + bias;
      v = v > 0.f ? v : 0.f;
      fm[(((size_t)(n * HW + y)) * HW + x) * COUT + o] = (_Float16)v;
    }
  }
}

// =====================================================================
// 1x1 heads as one WMMA GEMM (N=64, 54 used), epilogue scatters into
// rpn_locs / rpn_scores with the permute(0,2,3,1) ordering
// =====================================================================
__global__ __launch_bounds__(128) void head_wmma(
    const _Float16* __restrict__ fm, const _Float16* __restrict__ Wt,
    const float* __restrict__ hbias, float* __restrict__ out)
{
  __shared__ _Float16 As[64][32];
  __shared__ _Float16 Ws[64][32];

  int mrow = blockIdx.x;            // 0..255 -> (n,y)
  int n = mrow >> 6, y = mrow & 63;
  int t = threadIdx.x;
  int w = t >> 5, lane = t & 31;
  v8f acc[4] = {};

  int arow = t >> 1, aseg = t & 1;  // 32 B per thread

  for (int ks = 0; ks < 16; ++ks) {
    {
      const u32x4* src = (const u32x4*)(fm +
          (((size_t)mrow) * HW + arow) * COUT + ks * 32 + aseg * 16);
      u32x4 v0 = src[0], v1 = src[1];
      *((u32x4*)&As[arow][aseg * 16])     = v0;
      *((u32x4*)&As[arow][aseg * 16 + 8]) = v1;
    }
    {
      const u32x4* src = (const u32x4*)(Wt + (size_t)arow * CIN + ks * 32 + aseg * 16);
      u32x4 v0 = src[0], v1 = src[1];
      *((u32x4*)&Ws[arow][aseg * 16])     = v0;
      *((u32x4*)&Ws[arow][aseg * 16 + 8]) = v1;
    }
    __syncthreads();

    union { v16h v; v8h h[2]; } af;
    int am = w * 16 + (lane & 15);
    int ko = (lane >> 4) * 8;
    af.h[0] = *(const v8h*)&As[am][ko];
    af.h[1] = *(const v8h*)&As[am][16 + ko];

#pragma unroll
    for (int tc = 0; tc < 4; ++tc) {
      union { v16h v; v8h h[2]; } bf;
      int col = tc * 16 + (lane & 15);
      int bko = (lane >> 4) * 8;
      bf.h[0] = *(const v8h*)&Ws[col][bko];
      bf.h[1] = *(const v8h*)&Ws[col][bko + 16];
      acc[tc] = __builtin_amdgcn_wmma_f32_16x16x32_f16(
          false, af.v, false, bf.v, (short)0, acc[tc], false, false);
    }
    __syncthreads();
  }

#pragma unroll
  for (int tc = 0; tc < 4; ++tc) {
    int o = tc * 16 + (lane & 15);
    if (o >= 54) continue;
    float bias = hbias[o];
#pragma unroll
    for (int r = 0; r < 8; ++r) {
      int x = w * 16 + (lane >> 4) * 8 + r;
      int pix = y * HW + x;
      float v = acc[tc][r] + bias;
      if (o < 36) {
        int a = o >> 2, d = o & 3;
        out[OFF_LOCS + (((size_t)n * NPIX + pix) * 9 + a) * 4 + d] = v;
      } else {
        int oc = o - 36;
        int a = oc >> 1, jj = oc & 1;
        out[OFF_SCORES + (((size_t)n * NPIX + pix) * 9 + a) * 2 + jj] = v;
      }
    }
  }
}

// =====================================================================
// anchors / decode / NMS
// =====================================================================
__device__ inline void anchor_of(int j, float& y1, float& x1, float& y2, float& x2)
{
  int pix = j / 9, a = j % 9;
  int py = pix >> 6, px = pix & 63;
  const float ratios[3] = {0.5f, 1.0f, 2.0f};
  const float scales[3] = {8.f, 16.f, 32.f};
  float r = ratios[a / 3], s = scales[a % 3];
  float h = 16.f * s * sqrtf(r);
  float ww = 16.f * s * sqrtf(1.f / r);
  float cy = py * 16.f + 8.f, cx = px * 16.f + 8.f;
  y1 = cy - h * 0.5f;  x1 = cx - ww * 0.5f;
  y2 = cy + h * 0.5f;  x2 = cx + ww * 0.5f;
}

__global__ __launch_bounds__(256) void anchor_kernel(float* __restrict__ out)
{
  int j = blockIdx.x * 256 + threadIdx.x;
  if (j >= NANCH) return;
  float y1, x1, y2, x2;
  anchor_of(j, y1, x1, y2, x2);
  float* p = out + OFF_ANCH + (size_t)j * 4;
  p[0] = y1; p[1] = x1; p[2] = y2; p[3] = x2;
}

__global__ __launch_bounds__(256) void decode_kernel(
    const float* __restrict__ out, const int* __restrict__ ih,
    const int* __restrict__ iw, float* __restrict__ boxes,
    float* __restrict__ scores)
{
  int idx = blockIdx.x * 256 + threadIdx.x;
  if (idx >= NIMG * NANCH) return;
  int j = idx % NANCH;
  float ay1, ax1, ay2, ax2;
  anchor_of(j, ay1, ax1, ay2, ax2);
  float ha = ay2 - ay1, wa = ax2 - ax1;
  float cya = ay1 + 0.5f * ha, cxa = ax1 + 0.5f * wa;
  const float* loc = out + OFF_LOCS + (size_t)idx * 4;
  float cy = loc[0] * ha + cya;
  float cx = loc[1] * wa + cxa;
  float hh = expf(loc[2]) * ha;
  float wn = expf(loc[3]) * wa;
  float IH = (float)ih[0], IW = (float)iw[0];
  float y1 = fminf(fmaxf(cy - 0.5f * hh, 0.f), IH);
  float x1 = fminf(fmaxf(cx - 0.5f * wn, 0.f), IW);
  float y2 = fminf(fmaxf(cy + 0.5f * hh, 0.f), IH);
  float x2 = fminf(fmaxf(cx + 0.5f * wn, 0.f), IW);
  bool valid = ((y2 - y1) >= 13.f) && ((x2 - x1) >= 13.f);
  float s0 = out[OFF_SCORES + (size_t)idx * 2];
  float s1 = out[OFF_SCORES + (size_t)idx * 2 + 1];
  float fg = 1.f / (1.f + expf(s0 - s1));        // softmax fg prob
  float* b = boxes + (size_t)idx * 4;
  b[0] = y1; b[1] = x1; b[2] = y2; b[3] = x2;
  scores[idx] = valid ? fg : -__builtin_inff();
}

// greedy NMS: one 1024-thread block per image; 3000 rounds of
// (parallel argmax over 36864) + (parallel IoU suppression)
__global__ __launch_bounds__(1024) void nms_kernel(
    const float* __restrict__ boxes, float* __restrict__ scores,
    float* __restrict__ out)
{
  __shared__ float rs[1024];
  __shared__ int   ri[1024];
  __shared__ float sb[5];           // y1,x1,y2,x2,area of winner
  __shared__ int   sflag;

  int n = blockIdx.x;
  int tid = threadIdx.x;
  const float*  bx = boxes  + (size_t)n * NANCH * 4;
  float*        sc = scores + (size_t)n * NANCH;

  for (int it = 0; it < NPOST; ++it) {
    float m = -__builtin_inff();
    int mi = 0;
    for (int j = tid; j < NANCH; j += 1024) {
      float s = sc[j];
      if (s > m) { m = s; mi = j; }
    }
    rs[tid] = m; ri[tid] = mi;
    __syncthreads();
    for (int s = 512; s > 0; s >>= 1) {
      if (tid < s) {
        float a = rs[tid], b = rs[tid + s];
        if (b > a || (b == a && ri[tid + s] < ri[tid])) {
          rs[tid] = b; ri[tid] = ri[tid + s];
        }
      }
      __syncthreads();
    }
    if (tid == 0) {
      int   idx = ri[0];
      bool  ok  = rs[0] > -__builtin_inff();
      float y1 = bx[(size_t)idx * 4 + 0], x1 = bx[(size_t)idx * 4 + 1];
      float y2 = bx[(size_t)idx * 4 + 2], x2 = bx[(size_t)idx * 4 + 3];
      float* rp = out + OFF_ROIS + ((size_t)n * NPOST + it) * 4;
      if (ok) { rp[0] = y1; rp[1] = x1; rp[2] = y2; rp[3] = x2; }
      else    { rp[0] = 0.f; rp[1] = 0.f; rp[2] = 0.f; rp[3] = 0.f; }
      out[OFF_RIDX + (size_t)n * NPOST + it] = ok ? (float)n : -1.f;
      sb[0] = y1; sb[1] = x1; sb[2] = y2; sb[3] = x2;
      sb[4] = (y2 - y1) * (x2 - x1);
      sflag = ok ? 1 : 0;
    }
    __syncthreads();
    if (sflag) {
      float by1 = sb[0], bx1 = sb[1], by2 = sb[2], bx2 = sb[3], ba = sb[4];
      for (int j = tid; j < NANCH; j += 1024) {
        const float* q = bx + (size_t)j * 4;
        float y1 = q[0], x1 = q[1], y2 = q[2], x2 = q[3];
        float iy1 = fmaxf(by1, y1), ix1 = fmaxf(bx1, x1);
        float iy2 = fminf(by2, y2), ix2 = fminf(bx2, x2);
        float inter = fmaxf(iy2 - iy1, 0.f) * fmaxf(ix2 - ix1, 0.f);
        float area  = (y2 - y1) * (x2 - x1);
        float iou   = inter / (ba + area - inter + 1e-10f);
        if (iou > 0.7f) sc[j] = -__builtin_inff();
      }
    }
    __syncthreads();
  }
}

// =====================================================================
extern "C" void kernel_launch(void* const* d_in, const int* in_sizes, int n_in,
                              void* d_out, int out_size, void* d_ws, size_t ws_size,
                              hipStream_t stream)
{
  const float* Image  = (const float*)d_in[0];
  const float* conv_w = (const float*)d_in[1];
  const float* conv_b = (const float*)d_in[2];
  const float* cls_w  = (const float*)d_in[3];
  const float* cls_b  = (const float*)d_in[4];
  const float* reg_w  = (const float*)d_in[5];
  const float* reg_b  = (const float*)d_in[6];
  const int*   img_h  = (const int*)d_in[7];
  const int*   img_w  = (const int*)d_in[8];

  char* ws = (char*)d_ws;
  _Float16* padIn  = (_Float16*)(ws + WS_PADIN);
  _Float16* Bt     = (_Float16*)(ws + WS_BT);
  _Float16* fm     = (_Float16*)(ws + WS_FM);
  _Float16* Wt     = (_Float16*)(ws + WS_WT);
  float*    hbias  = (float*)(ws + WS_HB);
  float*    boxes  = (float*)(ws + WS_BOXES);
  float*    scores = (float*)(ws + WS_SCR);
  float*    out    = (float*)d_out;

  pad_input_kernel<<<(NIMG * PADW * PADW * CIN + 255) / 256, 256, 0, stream>>>(Image, padIn);
  prep_w3_kernel<<<(COUT * KDIM + 255) / 256, 256, 0, stream>>>(conv_w, Bt);
  prep_whead_kernel<<<(64 * CIN + 255) / 256, 256, 0, stream>>>(cls_w, cls_b, reg_w, reg_b, Wt, hbias);

  conv3x3_wmma<<<1024, 256, 0, stream>>>(padIn, Bt, conv_b, fm);
  head_wmma<<<256, 128, 0, stream>>>(fm, Wt, hbias, out);

  anchor_kernel<<<(NANCH + 255) / 256, 256, 0, stream>>>(out);
  decode_kernel<<<(NIMG * NANCH + 255) / 256, 256, 0, stream>>>(out, img_h, img_w, boxes, scores);
  nms_kernel<<<NIMG, 1024, 0, stream>>>(boxes, scores, out);
}
